// sLSTM_57286273794822
// MI455X (gfx1250) — matrix operand order repaired and checked by reading, not compiled, not run on hardware
//
#include <hip/hip_runtime.h>
#include <stdint.h>

// ---------------------------------------------------------------------------
// sLSTM for MI455X (gfx1250, wave32, WMMA + Tensor Data Mover), PERSISTENT.
// One launch: 64 blocks (one per WGP, co-resident) loop over T=1024 steps x
// 2 layers; grid-wide ticket barrier replaces 2048 kernel launches.
// Per layer-step GEMM: gates[64,4096] = [x_in | h][64,2048] @ Wcat^T[2048,4096]
//  - activations DMA'd to LDS per block via tensor_load_to_lds (TDM) with
//    descriptor padding (4 DW per 256 DW) to break LDS bank conflicts
//  - carry state (c,n,m) lives in LDS for the whole sequence (24 KB/block)
//  - weights stream from global (read once per step; L2-resident, 32 MB bf16)
//  - v_wmma_f32_16x16x32_bf16, fp32 accum, fused stabilized-gate update
// ---------------------------------------------------------------------------

typedef __attribute__((ext_vector_type(16))) __bf16   v16bf;
typedef __attribute__((ext_vector_type(8)))  float    v8f;
typedef __attribute__((ext_vector_type(4)))  uint32_t u32x4;
typedef __attribute__((ext_vector_type(8)))  uint32_t u32x8;
typedef __attribute__((ext_vector_type(2)))  uint32_t u32x2;
typedef __attribute__((ext_vector_type(4)))  float    f32x4;

// LDS layout (dynamic shared only, so offsets start at 0 for TDM lds_addr):
//  [0, 133120)        A region 0 (64 rows x 2048B, +16B pad per 1024B)
//  [133120, 266240)   A region 1 (same shape)
//  [266240, 283648)   gate staging float[4][64][17] (padded rows)
//  [283648, 308224)   carry state float[2 layers][c|n|m][64*16]
#define LDS_A_REGION_BYTES 133120u
#define LDS_GATES_OFF      266240u
#define LDS_STATE_OFF      283648u
#define LDS_TOTAL_BYTES    308224u
#define LDS_ROW_STRIDE     2080      // 2048B data + 2x16B pad per row
#define LDS_HALF_OFF       1040      // second 512-elem half: 1024B data + 16B pad
#define LDS_TILE16_OFF     (16 * LDS_ROW_STRIDE)   // +33280, fits DS 16-bit imm
#define GRID_BLOCKS        64

__device__ __forceinline__ uint16_t f2bf(float f) {
    uint32_t u = __float_as_uint(f);
    uint32_t r = u + 0x7FFFu + ((u >> 16) & 1u);   // round-to-nearest-even
    return (uint16_t)(r >> 16);
}

// ---------------------------------------------------------------------------
// TDM: 2D tile load Global -> LDS. Descriptor per CDNA5 ISA.
// group0: [1:0]=count=1, [63:32]=lds_addr, [120:64]=global_addr, [127:126]=2
// group1: [17:16]=data_size(1=2B), [20]=pad_en, [24:22]=pad_interval(7=256DW),
//         [31:25]=pad_amount(3=4DW), [79:48]=tensor_dim0, [111:80]=tensor_dim1,
//         [127:112]=tile_dim0, [143:128]=tile_dim1, [207:160]=dim0_stride
// ---------------------------------------------------------------------------
__device__ __forceinline__ void tdm_load_2d(uint32_t lds_off, const void* gptr,
                                            uint32_t rows, uint32_t cols,
                                            uint64_t row_stride_elems) {
    const uint64_t ga = (uint64_t)gptr;
    u32x4 g0;
    g0[0] = 1u;                                            // count=1, user desc
    g0[1] = lds_off;
    g0[2] = (uint32_t)ga;
    g0[3] = ((uint32_t)(ga >> 32) & 0x01FFFFFFu) | (2u << 30);  // type=2 (image)
    u32x8 g1;
    g1[0] = 0x00010000u | (1u << 20) | (7u << 22) | (3u << 25); // 2B, pad 4DW/256DW
    g1[1] = (cols & 0xFFFFu) << 16;                        // tensor_dim0 lo
    g1[2] = ((cols >> 16) & 0xFFFFu) | ((rows & 0xFFFFu) << 16); // dim0 hi | dim1 lo
    g1[3] = ((rows >> 16) & 0xFFFFu) | ((cols & 0xFFFFu) << 16); // dim1 hi | tile_dim0
    g1[4] = rows & 0xFFFFu;                                // tile_dim1 (tile_dim2=0)
    g1[5] = (uint32_t)row_stride_elems;                    // tensor_dim0_stride lo
    g1[6] = (uint32_t)(row_stride_elems >> 32) & 0xFFFFu;  // dim0_stride hi
    g1[7] = 0u;                                            // dim1_stride unused (2D)
    asm volatile("tensor_load_to_lds %0, %1" :: "s"(g0), "s"(g1) : "memory");
}

// Grid-wide barrier: monotonic ticket counter (no reset races). Leader thread
// arrives with device-scope release, spins with acquire + s_sleep backoff.
__device__ __forceinline__ void grid_sync(uint32_t* bar) {
    __syncthreads();
    if (threadIdx.x == 0) {
        __threadfence();
        uint32_t ticket = __hip_atomic_fetch_add(bar, 1u, __ATOMIC_ACQ_REL,
                                                 __HIP_MEMORY_SCOPE_AGENT);
        uint32_t target = ((ticket / GRID_BLOCKS) + 1u) * GRID_BLOCKS;
        while (__hip_atomic_load(bar, __ATOMIC_ACQUIRE,
                                 __HIP_MEMORY_SCOPE_AGENT) < target) {
            __builtin_amdgcn_s_sleep(2);
        }
        __threadfence();
    }
    __syncthreads();
}

// A fragment (16x32 bf16 MxK) from padded LDS: two b128 at +0 / +32.
__device__ __forceinline__ v16bf load_frag_a_lds(const char* p) {
    union { u32x4 q[2]; v16bf v; } f;
    f.q[0] = *reinterpret_cast<const u32x4*>(p);
    f.q[1] = *reinterpret_cast<const u32x4*>(p + 32);
    return f.v;
}

// B fragment (32x16 bf16 KxN) for gates = xh @ W^T: B[k][n] = W[n][k].
__device__ __forceinline__ v16bf load_frag_b(const uint16_t* __restrict__ row,
                                             int kk, int hi) {
    union { u32x4 q[2]; v16bf v; } f;
    const uint16_t* p = row + kk + hi * 8;
    f.q[0] = *reinterpret_cast<const u32x4*>(p);
    f.q[1] = *reinterpret_cast<const u32x4*>(p + 16);
    return f.v;
}

// 512-element K half: 16 K-steps, 4 M-tiles sharing one B fragment per step.
// The second anchor's offset is laundered through an empty asm so the
// compiler cannot re-associate baseB = baseA + const (which previously
// collapsed both anchors into one register + negative v_adds + WAR v_nops).
// Both anchors then serve all loads via positive 16-bit DS immediates.
__device__ __forceinline__ void khalf(const char* __restrict__ baseA,
                                      const uint16_t* __restrict__ wrowk, int hi,
                                      v8f& c0, v8f& c1, v8f& c2, v8f& c3) {
    int offB = 2 * LDS_TILE16_OFF;          // +66560
    asm("" : "+v"(offB));                   // opaque: forces a real 2nd anchor
    const char* baseB = baseA + offB;
    #pragma unroll 4
    for (int kk = 0; kk < 512; kk += 32) {
        const int bo = kk * 2;
        v16bf bf = load_frag_b(wrowk, kk, hi);
        v16bf a0 = load_frag_a_lds(baseA + bo);
        v16bf a1 = load_frag_a_lds(baseA + LDS_TILE16_OFF + bo);
        v16bf a2 = load_frag_a_lds(baseB + bo);
        v16bf a3 = load_frag_a_lds(baseB + LDS_TILE16_OFF + bo);
        c0 = __builtin_amdgcn_wmma_f32_16x16x32_bf16(false, a0, false, bf,
                                                     (short)0, c0, false, false);
        c1 = __builtin_amdgcn_wmma_f32_16x16x32_bf16(false, a1, false, bf,
                                                     (short)0, c1, false, false);
        c2 = __builtin_amdgcn_wmma_f32_16x16x32_bf16(false, a2, false, bf,
                                                     (short)0, c2, false, false);
        c3 = __builtin_amdgcn_wmma_f32_16x16x32_bf16(false, a3, false, bf,
                                                     (short)0, c3, false, false);
    }
}

// One layer of one time step (called inside the persistent loop).
__device__ __forceinline__ void layer_step(
    const uint16_t* __restrict__ A0, int64_t strideA0,   // input half of xh
    const uint16_t* __restrict__ A1,                     // h half (stride 1024)
    const uint16_t* __restrict__ wrow,                   // this wave's weight rows
    const float*    __restrict__ biasL,                  // [4H] this layer
    float* __restrict__ stateL,                          // LDS c|n|m, 3*1024 floats
    uint16_t* __restrict__ hw,                           // new h, bf16 [64][1024]
    float* __restrict__ outT,                            // out + t*H or null
    float* __restrict__ hAllL,                           // h_all + layer*BH or null
    int gate, int hi, int lm, int nb)
{
    extern __shared__ char smem[];
    float (*g_lds)[64][17] = (float (*)[64][17])(smem + LDS_GATES_OFF);

    // Wave 0 DMAs both activation halves into LDS; TENSORcnt wait, then the
    // workgroup barrier publishes LDS to the other waves.
    if (threadIdx.x == 0) {
        tdm_load_2d(0u,                 A0, 64u, 1024u, (uint64_t)strideA0);
        tdm_load_2d(LDS_A_REGION_BYTES, A1, 64u, 1024u, 1024u);
        __builtin_amdgcn_s_wait_tensorcnt(0);
    }
    __syncthreads();

    const char* anchor = smem + (size_t)lm * LDS_ROW_STRIDE + hi * 16;
    v8f c0 = {}, c1 = {}, c2 = {}, c3 = {};
    khalf(anchor,                                     wrow,        hi, c0, c1, c2, c3);
    khalf(anchor + LDS_HALF_OFF,                      wrow + 512,  hi, c0, c1, c2, c3);
    khalf(anchor + LDS_A_REGION_BYTES,                wrow + 1024, hi, c0, c1, c2, c3);
    khalf(anchor + LDS_A_REGION_BYTES + LDS_HALF_OFF, wrow + 1536, hi, c0, c1, c2, c3);

    // C/D layout: element r of lane -> M = mtile*16 + r + hi*8, N = lm.
    #pragma unroll
    for (int r = 0; r < 8; ++r) {
        g_lds[gate][ 0 + r + hi * 8][lm] = c0[r];
        g_lds[gate][16 + r + hi * 8][lm] = c1[r];
        g_lds[gate][32 + r + hi * 8][lm] = c2[r];
        g_lds[gate][48 + r + hi * 8][lm] = c3[r];
    }
    __syncthreads();

    // Fused stabilized-gate update; carry state (c,n,m) stays in LDS.
    for (int i = threadIdx.x; i < 64 * 16; i += 128) {
        const int b  = i >> 4;
        const int jj = i & 15;
        const int j  = nb + jj;
        float zt = g_lds[0][b][jj] + biasL[j];
        float it = g_lds[1][b][jj] + biasL[1024 + j];
        float ft = g_lds[2][b][jj] + biasL[2048 + j];
        float ot = g_lds[3][b][jj] + biasL[3072 + j];
        float c = stateL[i], n = stateL[1024 + i], m = stateL[2048 + i];
        float mn = fmaxf(ft + m, it);             // m_t = max(log f + m, log i)
        float ip = __expf(it - mn);
        float fp = __expf(ft + m - mn);
        float z  = tanhf(zt);
        float o  = 1.0f / (1.0f + __expf(-ot));
        float cn = fp * c + ip * z;
        float nn = fp * n + ip;
        float h  = o * (cn / nn);
        stateL[i] = cn; stateL[1024 + i] = nn; stateL[2048 + i] = mn;
        const int off = b * 1024 + j;
        hw[off] = f2bf(h);
        if (outT)  outT[(int64_t)b * 1048576 + j] = h;   // outputs[b][t][j]
        if (hAllL) hAllL[off] = h;                       // final-step h_all
    }
}

// Persistent kernel: 64 blocks x 128 threads (4 waves; wave id == gate id).
// Block owns hidden columns [nb, nb+16) of both layers for the whole sequence.
__global__ void __launch_bounds__(128)
slstm_persistent(const uint16_t* __restrict__ Xbf,    // [B][T][D] bf16
                 const uint16_t* __restrict__ Wcat,   // [L][4096][2048] bf16
                 const float*    __restrict__ bias,   // [L][4H]
                 uint16_t* __restrict__ hbf,          // 4 x [64][1024] bf16 (dbl buf)
                 float* __restrict__ out,             // [B][T][H]
                 float* __restrict__ hAll, float* __restrict__ cAll,
                 float* __restrict__ nAll, float* __restrict__ mAll,
                 uint32_t* __restrict__ bar)
{
    extern __shared__ char smem[];
    float* stateLds = (float*)(smem + LDS_STATE_OFF);  // [2][c|n|m][1024]

    const int lane = threadIdx.x & 31;
    const int gate = threadIdx.x >> 5;
    const int hi   = lane >> 4;
    const int lm   = lane & 15;
    const int nb   = blockIdx.x * 16;

    // Init carry state in LDS: c=0, n=1, m=0 for both layers.
    for (int i = threadIdx.x; i < 6144; i += 128)
        stateLds[i] = (((i >> 10) % 3) == 1) ? 1.0f : 0.0f;
    __syncthreads();

    const uint16_t* w0 = Wcat + (int64_t)(gate * 1024 + nb + lm) * 2048;
    const uint16_t* w1 = w0 + (int64_t)4096 * 2048;

    for (int t = 0; t < 1024; ++t) {
        const int p = t & 1;
        uint16_t* h0r = hbf + p * 65536;
        uint16_t* h0w = hbf + (1 - p) * 65536;
        uint16_t* h1r = hbf + 131072 + p * 65536;
        uint16_t* h1w = hbf + 131072 + (1 - p) * 65536;
        const bool last = (t == 1023);

        // layer 0: input = x[:,t,:] (row stride T*D), recurrent = h0r
        layer_step(Xbf + (int64_t)t * 1024, (int64_t)1048576, h0r,
                   w0, bias, stateLds, h0w,
                   (float*)nullptr, last ? hAll : (float*)nullptr,
                   gate, hi, lm, nb);
        grid_sync(bar);   // publish h0w to all blocks

        // layer 1: input = new h0, recurrent = h1r; writes outputs[:,t,:]
        layer_step(h0w, (int64_t)1024, h1r,
                   w1, bias + 4096, stateLds + 3072, h1w,
                   out + (int64_t)t * 1024, last ? hAll + 65536 : (float*)nullptr,
                   gate, hi, lm, nb);
        grid_sync(bar);   // publish h1w; also fences LDS A-region reuse
    }

    // Final carry state: LDS -> d_out tail (c_all, n_all, m_all).
    for (int i = threadIdx.x; i < 1024; i += 128) {
        const int b = i >> 4, jj = i & 15, j = nb + jj;
        const int off = b * 1024 + j;
        cAll[off]         = stateLds[i];
        nAll[off]         = stateLds[1024 + i];
        mAll[off]         = stateLds[2048 + i];
        cAll[65536 + off] = stateLds[3072 + i];
        nAll[65536 + off] = stateLds[4096 + i];
        mAll[65536 + off] = stateLds[5120 + i];
    }
}

// Zero the bf16 h double buffers and the grid barrier counter.
__global__ void init_state(uint16_t* __restrict__ hbf, uint32_t* __restrict__ bar) {
    const int i = blockIdx.x * blockDim.x + threadIdx.x;  // < 4*B*H = 262144
    hbf[i] = 0;
    if (i == 0) bar[0] = 0;
}

// Wcat[l][row][0:1024]=W_ih[l][row][:], [1024:2048]=W_hh[l][row][:] (bf16)
__global__ void conv_w(const float* __restrict__ Wi, const float* __restrict__ Wh,
                       uint16_t* __restrict__ Wc) {
    const int64_t i   = (int64_t)blockIdx.x * 256 + threadIdx.x;  // < 2*4096*2048
    const int64_t row = i >> 11;
    const int     k   = (int)(i & 2047);
    const float   v   = (k < 1024) ? Wi[row * 1024 + k] : Wh[row * 1024 + (k - 1024)];
    Wc[i] = f2bf(v);
}

// x fp32 -> bf16, vectorized x4 (b128 in, b64 out)
__global__ void conv_x(const float* __restrict__ x, uint16_t* __restrict__ xb) {
    const int64_t i = ((int64_t)blockIdx.x * 256 + threadIdx.x) * 4;
    f32x4 v = *reinterpret_cast<const f32x4*>(x + i);
    u32x2 p;
    p.x = (uint32_t)f2bf(v.x) | ((uint32_t)f2bf(v.y) << 16);
    p.y = (uint32_t)f2bf(v.z) | ((uint32_t)f2bf(v.w) << 16);
    *reinterpret_cast<u32x2*>(xb + i) = p;
}

extern "C" void kernel_launch(void* const* d_in, const int* in_sizes, int n_in,
                              void* d_out, int out_size, void* d_ws, size_t ws_size,
                              hipStream_t stream) {
    const float* x    = (const float*)d_in[0];   // [B,T,D]
    const float* W_ih = (const float*)d_in[1];   // [L,4H,D]
    const float* W_hh = (const float*)d_in[2];   // [L,4H,H]
    const float* bias = (const float*)d_in[3];   // [L,4H]
    float* out = (float*)d_out;

    constexpr int     B = 64, T = 1024, D = 1024, H = 1024, L = 2;
    constexpr int     BH  = B * H;               // 65536
    constexpr int64_t BTH = (int64_t)B * T * H;  // 67108864

    // Allow >64KB dynamic LDS (host-side attr; not a stream op, capture-safe).
    hipFuncSetAttribute((const void*)slstm_persistent,
                        hipFuncAttributeMaxDynamicSharedMemorySize,
                        (int)LDS_TOTAL_BYTES);

    // Workspace carve-out (256B aligned): Xbf 128MB, Wcat 32MB, hbf 512KB, bar.
    char*  ws  = (char*)d_ws;
    size_t off = 0;
    auto carve = [&](size_t bytes) {
        void* p = ws + off;
        off = (off + bytes + 255) & ~(size_t)255;
        return p;
    };
    uint16_t* Xbf  = (uint16_t*)carve((size_t)B * T * D * 2);
    uint16_t* Wcat = (uint16_t*)carve((size_t)L * 4 * H * 2048 * 2);
    uint16_t* hbf  = (uint16_t*)carve((size_t)4 * BH * 2);   // h0{p0,p1}, h1{p0,p1}
    uint32_t* bar  = (uint32_t*)carve(256);

    // Carry state lives in d_out tail: [h_all][c_all][n_all][m_all], each L*B*H.
    float* hAll = out + BTH;
    float* cAll = hAll + (int64_t)L * BH;
    float* nAll = cAll + (int64_t)L * BH;
    float* mAll = nAll + (int64_t)L * BH;

    init_state<<<1024, 256, 0, stream>>>(hbf, bar);
    conv_w<<<65536, 256, 0, stream>>>(W_ih, W_hh, Wcat);
    conv_x<<<65536, 256, 0, stream>>>(x, Xbf);

    slstm_persistent<<<GRID_BLOCKS, 128, LDS_TOTAL_BYTES, stream>>>(
        Xbf, Wcat, bias, hbf, out, hAll, cAll, nAll, mAll, bar);
}
